// CrossAttn_34935263986214
// MI455X (gfx1250) — matrix operand order repaired and checked by reading, not compiled, not run on hardware
//
#include <hip/hip_runtime.h>
#include <hip/hip_bf16.h>

#define Dm 1024
#define Hh 16
#define DH 64
#define Bsz 4
#define Mlen 1024
#define Nlen 2048
#define LN_EPS 1e-5f
#define LOG2E 1.4426950408889634f

typedef _Float16 half16 __attribute__((ext_vector_type(16)));
typedef _Float16 half8  __attribute__((ext_vector_type(8)));
typedef float    f32x8  __attribute__((ext_vector_type(8)));
typedef float    f32x4  __attribute__((ext_vector_type(4)));

union H8U { half8 h; unsigned u[4]; };
union H16U { half16 h; unsigned u[8]; };

// A-matrix 16x32 f16 fragment: per lane-half 16 CONTIGUOUS K values (K = 16*hf + j).
// caller passes p already offset by +8*hf; chunks at +8*hf and +8*hf+8.
__device__ __forceinline__ half16 load_fragA(const _Float16* p) {
  half8 lo = *(const half8*)(p + 8 * ((threadIdx.x >> 4) & 1));
  half8 hi = *(const half8*)(p + 8 * ((threadIdx.x >> 4) & 1) + 8);
  half16 r;
#pragma unroll
  for (int i = 0; i < 8; ++i) { r[i] = lo[i]; r[i + 8] = hi[i]; }
  return r;
}

// B-matrix 32x16 f16 fragment: per lane-half 16 CONTIGUOUS K values (K = 16*hf + j).
__device__ __forceinline__ half16 load_fragB(const _Float16* p, int hf) {
  return *(const half16*)(p + 16 * hf);
}

__device__ __forceinline__ f32x8 wmma_f16(half16 a, half16 b, f32x8 c) {
  return __builtin_amdgcn_wmma_f32_16x16x32_f16(false, a, false, b, (short)0, c, false, false);
}

// ---------------- fp32 -> f16 conversion (8 elems/lane) ----------------
__global__ void cvt_f16_kernel(const float* __restrict__ src, _Float16* __restrict__ dst, int n) {
  int i = (blockIdx.x * blockDim.x + threadIdx.x) * 8;
  const int stride = gridDim.x * blockDim.x * 8;
  for (; i < n; i += stride) {
    f32x4 a = *(const f32x4*)(src + i);
    f32x4 b = *(const f32x4*)(src + i + 4);
    half8 h;
#pragma unroll
    for (int j = 0; j < 4; ++j) { h[j] = (_Float16)a[j]; h[j + 4] = (_Float16)b[j]; }
    *(half8*)(dst + i) = h;
  }
}

// ---------------- elementwise scale (gate bias -> log2 domain) ----------------
__global__ void scale_f32_kernel(const float* __restrict__ src, float* __restrict__ dst,
                                 int n, float s) {
  int i = blockIdx.x * blockDim.x + threadIdx.x;
  if (i < n) dst[i] = src[i] * s;
}

// ---------------- WMMA GEMM: Y = (X @ W^T + bias) * outScale, 64x64 block/wave ------
// MODE 0: f16 out row-major; MODE 1: f16 out transposed per batch vT[b][feat][n];
// MODE 2: f32 out row-major
template <int MODE>
__global__ __attribute__((amdgpu_waves_per_eu(2)))
void wmma_gemm_kernel(const _Float16* __restrict__ A,
                      const _Float16* __restrict__ W,
                      const float* __restrict__ bias,
                      void* __restrict__ out,
                      int rows, int nPerB, float outScale) {
  const int lane  = threadIdx.x & 31;
  const int wave  = threadIdx.x >> 5;
  const int wglob = blockIdx.x * (blockDim.x >> 5) + wave;
  const int colBlocks = Dm / 64;  // 16
  const int rowBlock  = wglob / colBlocks;
  const int colBlock  = wglob - rowBlock * colBlocks;
  if (rowBlock * 64 >= rows) return;

  const int hf = lane >> 4;
  const int l  = lane & 15;
  const int m0 = rowBlock * 64;
  const int n0 = colBlock * 64;

  const _Float16* ap = A + (size_t)(m0 + l) * Dm + hf * 8;  // A-frag addressing
  const _Float16* bp = W + (size_t)(n0 + l) * Dm;           // B-frag addressing

  f32x8 acc[4][4];
#pragma unroll
  for (int mi = 0; mi < 4; ++mi)
#pragma unroll
    for (int nj = 0; nj < 4; ++nj) acc[mi][nj] = (f32x8){};

  for (int k0 = 0; k0 < Dm; k0 += 32) {
    __builtin_prefetch(ap + k0 + 128, 0, 1);
    __builtin_prefetch(bp + k0 + 128, 0, 1);
    half16 af[4];
#pragma unroll
    for (int mi = 0; mi < 4; ++mi)
      af[mi] = load_fragA(ap + (size_t)mi * 16 * Dm + k0);
#pragma unroll
    for (int nj = 0; nj < 4; ++nj) {
      half16 bf = load_fragB(bp + (size_t)nj * 16 * Dm + k0, hf);
#pragma unroll
      for (int mi = 0; mi < 4; ++mi)
        acc[mi][nj] = wmma_f16(af[mi], bf, acc[mi][nj]);
    }
  }

  float bb[4];
#pragma unroll
  for (int nj = 0; nj < 4; ++nj) bb[nj] = bias ? bias[n0 + nj * 16 + l] : 0.f;

  if (MODE == 0) {
    _Float16* o = (_Float16*)out;
#pragma unroll
    for (int mi = 0; mi < 4; ++mi)
#pragma unroll
      for (int nj = 0; nj < 4; ++nj)
#pragma unroll
        for (int i = 0; i < 8; ++i)
          o[(size_t)(m0 + mi * 16 + i + 8 * hf) * Dm + n0 + nj * 16 + l] =
              (_Float16)((acc[mi][nj][i] + bb[nj]) * outScale);
  } else if (MODE == 1) {
    const int b    = m0 / nPerB;  // 64-row blocks never cross a batch (N % 64 == 0)
    const int nloc = m0 - b * nPerB;
#pragma unroll
    for (int mi = 0; mi < 4; ++mi)
#pragma unroll
      for (int nj = 0; nj < 4; ++nj) {
        half8 h;
#pragma unroll
        for (int i = 0; i < 8; ++i) h[i] = (_Float16)((acc[mi][nj][i] + bb[nj]) * outScale);
        _Float16* o = (_Float16*)out +
                      ((size_t)(b * Dm + n0 + nj * 16 + l)) * nPerB + nloc + mi * 16 + 8 * hf;
        *(half8*)o = h;
      }
  } else {
    float* o = (float*)out;
#pragma unroll
    for (int mi = 0; mi < 4; ++mi)
#pragma unroll
      for (int nj = 0; nj < 4; ++nj)
#pragma unroll
        for (int i = 0; i < 8; ++i)
          o[(size_t)(m0 + mi * 16 + i + 8 * hf) * Dm + n0 + nj * 16 + l] =
              (acc[mi][nj][i] + bb[nj]) * outScale;
  }
}

// ---------------- fused attention, TRANSPOSED formulation ----------------
// S^T = K q^T  (keys on VGPR axis, queries on lane axis), O^T = V^T P^T.
// * q is pre-scaled by 0.125*log2e at projection; gate bias pre-multiplied by log2e,
//   so P = 2^(st + g2): one v_pk_add + one raw v_exp_f32 per pair
//   (__builtin_amdgcn_exp2f avoids clang's denormal-range fixup; exponents are
//    bounded ~|15| by construction so the guard is unnecessary).
// * P^T B-fragment built with 8 dword shfl_xor(16) exchanges: no LDS, no barriers.
// * Softmax denominator: one scalar per lane + single shfl_xor(16) at the end.
// * No running max: logits bounded -> fp32 sum-of-exp is exact softmax.
// One wave = 32 query rows of one head.
__global__ __launch_bounds__(32) __attribute__((amdgpu_waves_per_eu(2)))
void attn_kernel(const _Float16* __restrict__ qh,  // [B*M, D] (pre-scaled)
                 const _Float16* __restrict__ kh,  // [B*N, D]
                 const _Float16* __restrict__ vT,  // [B, D, N]
                 const float* __restrict__ g2,     // [B, N] = lgb * log2e
                 _Float16* __restrict__ ctx)       // [B*M, D]
{
  const int lane = threadIdx.x & 31;
  const int hf   = lane >> 4;
  const int l    = lane & 15;
  const int bh = blockIdx.x;
  const int b  = bh / Hh;
  const int h  = bh - b * Hh;
  const int m0 = blockIdx.y * 32;

  // Q as B-fragments: column m = l, K = d (two 32-wide chunks of DH)
  half16 bq[2][2];
#pragma unroll
  for (int mt = 0; mt < 2; ++mt) {
    const _Float16* qp = qh + (size_t)(b * Mlen + m0 + mt * 16 + l) * Dm + h * DH;
    bq[mt][0] = load_fragB(qp, hf);
    bq[mt][1] = load_fragB(qp + 32, hf);
  }

  const _Float16* kbase = kh + (size_t)(b * Nlen) * Dm + h * DH + hf * 8;
  const _Float16* vbase = vT + ((size_t)b * Dm + h * DH + l) * Nlen + hf * 8;
  const float* g2p = g2 + b * Nlen + 8 * hf;  // per-element gate bias (keys on VGPR axis)

  f32x8 o[2][4];
#pragma unroll
  for (int mt = 0; mt < 2; ++mt)
#pragma unroll
    for (int dt = 0; dt < 4; ++dt) o[mt][dt] = (f32x8){};
  float lsum = 0.f;

#pragma unroll 1
  for (int n0 = 0; n0 < Nlen; n0 += 32) {
    // K as A-fragments (2 key tiles x 2 d-chunks) and V fragments, loaded together
    const _Float16* kp0 = kbase + (size_t)(n0 + l) * Dm;
    const _Float16* kp1 = kbase + (size_t)(n0 + 16 + l) * Dm;
    const half16 ak00 = load_fragA(kp0);
    const half16 ak01 = load_fragA(kp0 + 32);
    const half16 ak10 = load_fragA(kp1);
    const half16 ak11 = load_fragA(kp1 + 32);
    const _Float16* vp = vbase + n0;
    half16 av[4];
#pragma unroll
    for (int dt = 0; dt < 4; ++dt)
      av[dt] = load_fragA(vp + (size_t)(dt * 16) * Nlen);
    // gate bias (log2 domain) for this lane's key slots: key = n0 + 16*nt + 8*hf + i
    const f32x8 g0v = *(const f32x8*)(g2p + n0);
    const f32x8 g1v = *(const f32x8*)(g2p + n0 + 16);

    half16 pb[2];
#pragma unroll
    for (int mt = 0; mt < 2; ++mt) {
      f32x8 st0 = {}, st1 = {};
      st0 = wmma_f16(ak00, bq[mt][0], st0);
      st0 = wmma_f16(ak01, bq[mt][1], st0);
      st1 = wmma_f16(ak10, bq[mt][0], st1);
      st1 = wmma_f16(ak11, bq[mt][1], st1);
      H8U pa, pc;
#pragma unroll
      for (int i = 0; i < 8; ++i) {
        const float p0 = __builtin_amdgcn_exp2f(st0[i] + g0v[i]);
        const float p1 = __builtin_amdgcn_exp2f(st1[i] + g1v[i]);
        lsum += p0 + p1;
        pa.h[i] = (_Float16)p0;
        pc.h[i] = (_Float16)p1;
      }
      // Build P^T B-fragment: lane-half hf needs contiguous keys 16*hf..16*hf+15.
      // Own regs hold keys {8hf..8hf+7} of each tile; partner (lane^16) holds the rest.
      H16U bfrag;
#pragma unroll
      for (int j = 0; j < 4; ++j) {
        const unsigned ax = __shfl_xor(pa.u[j], 16, 32);
        const unsigned cx = __shfl_xor(pc.u[j], 16, 32);
        bfrag.u[j]     = hf ? cx : pa.u[j];
        bfrag.u[j + 4] = hf ? pc.u[j] : ax;
      }
      pb[mt] = bfrag.h;
    }
    // O^T += V^T P^T : V fragments shared by both m-tiles
#pragma unroll
    for (int dt = 0; dt < 4; ++dt) {
      o[0][dt] = wmma_f16(av[dt], pb[0], o[0][dt]);
      o[1][dt] = wmma_f16(av[dt], pb[1], o[1][dt]);
    }
  }

  // softmax denominator: one scalar per lane (query m = l), pair-reduce across hf
  const float inv = 1.0f / (lsum + __shfl_xor(lsum, 16, 32));

  // store context: O^T element (vgpr i, lane) = O[m=l][d = dt*16 + i + 8hf]
  // -> 8 consecutive d per lane: one b128 store per (mt, dt)
#pragma unroll
  for (int mt = 0; mt < 2; ++mt) {
    _Float16* cp = ctx + (size_t)(b * Mlen + m0 + mt * 16 + l) * Dm + h * DH + 8 * hf;
#pragma unroll
    for (int dt = 0; dt < 4; ++dt) {
      half8 hh;
#pragma unroll
      for (int i = 0; i < 8; ++i) hh[i] = (_Float16)(o[mt][dt][i] * inv);
      *(half8*)(cp + dt * 16) = hh;
    }
  }
}

// ---------------- residual + LayerNorm ----------------
__global__ void ln_kernel(const float* __restrict__ Qin, const float* __restrict__ y,
                          const float* __restrict__ gamma, const float* __restrict__ beta,
                          float* __restrict__ out) {
  __shared__ float red[8];
  const int row = blockIdx.x;
  const int t = threadIdx.x;  // 256
  const float* qr = Qin + (size_t)row * Dm;
  const float* yr = y + (size_t)row * Dm;
  float x[4];
  float s = 0.f;
#pragma unroll
  for (int i = 0; i < 4; ++i) { int c = t + i * 256; x[i] = qr[c] + yr[c]; s += x[i]; }
#pragma unroll
  for (int m = 16; m >= 1; m >>= 1) s += __shfl_xor(s, m, 32);
  if ((t & 31) == 0) red[t >> 5] = s;
  __syncthreads();
  s = 0.f;
#pragma unroll
  for (int i = 0; i < 8; ++i) s += red[i];
  const float mu = s * (1.f / Dm);
  __syncthreads();
  float v = 0.f;
#pragma unroll
  for (int i = 0; i < 4; ++i) { float d = x[i] - mu; v += d * d; }
#pragma unroll
  for (int m = 16; m >= 1; m >>= 1) v += __shfl_xor(v, m, 32);
  if ((t & 31) == 0) red[t >> 5] = v;
  __syncthreads();
  v = 0.f;
#pragma unroll
  for (int i = 0; i < 8; ++i) v += red[i];
  const float rstd = rsqrtf(v * (1.f / Dm) + LN_EPS);
#pragma unroll
  for (int i = 0; i < 4; ++i) {
    int c = t + i * 256;
    out[(size_t)row * Dm + c] = gamma[c] * (x[i] - mu) * rstd + beta[c];
  }
}

extern "C" void kernel_launch(void* const* d_in, const int* in_sizes, int n_in,
                              void* d_out, int out_size, void* d_ws, size_t ws_size,
                              hipStream_t stream) {
  const float* Q     = (const float*)d_in[0];
  const float* KV    = (const float*)d_in[1];
  const float* lgb   = (const float*)d_in[2];
  const float* Wq    = (const float*)d_in[3];
  const float* bq    = (const float*)d_in[4];
  const float* Wk    = (const float*)d_in[5];
  const float* bk    = (const float*)d_in[6];
  const float* Wv    = (const float*)d_in[7];
  const float* bv    = (const float*)d_in[8];
  const float* Wo    = (const float*)d_in[9];
  const float* bo    = (const float*)d_in[10];
  const float* gamma = (const float*)d_in[11];
  const float* beta  = (const float*)d_in[12];

  char* ws = (char*)d_ws;
  size_t off = 0;
  auto alloc = [&](size_t bytes) { char* p = ws + off; off += (bytes + 255) & ~(size_t)255; return p; };
  _Float16* Qh   = (_Float16*)alloc((size_t)Bsz * Mlen * Dm * 2);
  _Float16* KVh  = (_Float16*)alloc((size_t)Bsz * Nlen * Dm * 2);
  _Float16* Wqh  = (_Float16*)alloc((size_t)Dm * Dm * 2);
  _Float16* Wkh  = (_Float16*)alloc((size_t)Dm * Dm * 2);
  _Float16* Wvh  = (_Float16*)alloc((size_t)Dm * Dm * 2);
  _Float16* Woh  = (_Float16*)alloc((size_t)Dm * Dm * 2);
  _Float16* qhd  = (_Float16*)alloc((size_t)Bsz * Mlen * Dm * 2);
  _Float16* khd  = (_Float16*)alloc((size_t)Bsz * Nlen * Dm * 2);
  _Float16* vTd  = (_Float16*)alloc((size_t)Bsz * Dm * Nlen * 2);
  _Float16* ctxh = (_Float16*)alloc((size_t)Bsz * Mlen * Dm * 2);
  float*    y32  = (float*)   alloc((size_t)Bsz * Mlen * Dm * 4);
  float*    g2   = (float*)   alloc((size_t)Bsz * Nlen * 4);

  auto cvt = [&](const float* s, _Float16* d, int n) {
    int blocks = (n / 8 + 255) / 256;
    if (blocks > 4096) blocks = 4096;
    cvt_f16_kernel<<<blocks, 256, 0, stream>>>(s, d, n);
  };
  cvt(Q,  Qh,  Bsz * Mlen * Dm);
  cvt(KV, KVh, Bsz * Nlen * Dm);
  cvt(Wq, Wqh, Dm * Dm);
  cvt(Wk, Wkh, Dm * Dm);
  cvt(Wv, Wvh, Dm * Dm);
  cvt(Wo, Woh, Dm * Dm);
  // gate bias -> log2 domain
  scale_f32_kernel<<<(Bsz * Nlen + 255) / 256, 256, 0, stream>>>(lgb, g2, Bsz * Nlen, LOG2E);

  // q = (Qh @ Wq^T + bq) * (0.125 * log2e)   (rows = B*M)
  {
    int rows = Bsz * Mlen;
    int waves = (rows / 64) * (Dm / 64);
    wmma_gemm_kernel<0><<<waves / 4, 128, 0, stream>>>(Qh, Wqh, bq, qhd, rows, 0,
                                                       0.125f * LOG2E);
  }
  // k = KVh @ Wk^T + bk  (rows = B*N)
  {
    int rows = Bsz * Nlen;
    int waves = (rows / 64) * (Dm / 64);
    wmma_gemm_kernel<0><<<waves / 4, 128, 0, stream>>>(KVh, Wkh, bk, khd, rows, 0, 1.0f);
  }
  // vT[b][feat][n] = (KVh @ Wv^T + bv) transposed per batch
  {
    int rows = Bsz * Nlen;
    int waves = (rows / 64) * (Dm / 64);
    wmma_gemm_kernel<1><<<waves / 4, 128, 0, stream>>>(KVh, Wvh, bv, vTd, rows, Nlen, 1.0f);
  }
  // fused attention (32 query rows per wave, transposed flash loop)
  attn_kernel<<<dim3(Bsz * Hh, Mlen / 32), 32, 0, stream>>>(qhd, khd, vTd, g2, ctxh);
  // y = ctx @ Wo^T + bo (fp32 out)
  {
    int rows = Bsz * Mlen;
    int waves = (rows / 64) * (Dm / 64);
    wmma_gemm_kernel<2><<<waves / 4, 128, 0, stream>>>(ctxh, Woh, bo, y32, rows, 0, 1.0f);
  }
  // out = LayerNorm(Q + y) * gamma + beta
  ln_kernel<<<Bsz * Mlen, 256, 0, stream>>>(Q, y32, gamma, beta, (float*)d_out);
}
